// PartialFFTConv_52767968199448
// MI455X (gfx1250) — compile-verified, matrix operand order checked by reading, not run on hardware
//
#include <hip/hip_runtime.h>

// Causal depthwise FIR (128 taps) == reference's truncated-filter FFT conv.
//   y[b,h,l] = sum_{t<128} k[h,t] * x[b,h,l-t]
// GEMM mapping per 256-output tile (m,n in [0,16)):
//   y[l0+16m+n] = sum_{u'=0}^{159} xs[l0+16m+u'] * B[u',n],  B[u',n]=k[n-u'+128] (else 0)
// 5 chunks of K=32 -> v_wmma_f32_16x16x32_bf16; f32 accuracy via bf16 hi/lo
// truncation split: Ah*Bh + Ah*Bl + Al*Bh (lo exactly compensates hi truncation).
//
// Grid: one block per (head, L-half). Toeplitz filter staged once per block,
// B fragments live in registers across 8 batches x 8 tiles. Next batch's x
// segment is register-prefetched behind the current batch's WMMA work, and
// A fragments are double-buffered so chunk j+1's ds_load_b128s are in flight
// while chunk j's WMMAs execute (avoids s_wait_dscnt 0x0 serialization).

#define HD   768
#define LQ   4096
#define SEG  2048                 // L-split segment per block
#define TAPS 128
#define UPAD 160                  // padded contraction length (5 x 32)
#define BT_STRIDE 168             // LDS row stride (elems), bank-skewed
#define XS_LEN (128 + SEG + 32)   // 2208: causal/halo pad + segment + guard

typedef __attribute__((ext_vector_type(16))) __bf16        v16bf;
typedef __attribute__((ext_vector_type(8)))  float         v8f;
typedef __attribute__((ext_vector_type(4)))  unsigned int  u32x4;
typedef __attribute__((ext_vector_type(2)))  unsigned int  u32x2;
typedef __attribute__((ext_vector_type(4)))  float         f32x4;

union V16 {
    v16bf bf;
    u32x4 q[2];
};

// truncation bf16 split helpers (lo term compensates hi truncation exactly)
__device__ __forceinline__ unsigned int pack_hi16(float a, float b) {
    return (__float_as_uint(b) & 0xFFFF0000u) | (__float_as_uint(a) >> 16);
}
__device__ __forceinline__ float trunc_bf(float a) {
    return __uint_as_float(__float_as_uint(a) & 0xFFFF0000u);
}

__device__ __forceinline__ void stage8(f32x4 v0, f32x4 v1,
                                       unsigned short* dst_hi,
                                       unsigned short* dst_lo) {
    float f[8];
#pragma unroll
    for (int i = 0; i < 4; ++i) { f[i] = v0[i]; f[4 + i] = v1[i]; }
    u32x4 ph, pl;
#pragma unroll
    for (int p = 0; p < 4; ++p) {
        float a = f[2 * p], b = f[2 * p + 1];
        ph[p] = pack_hi16(a, b);
        pl[p] = pack_hi16(a - trunc_bf(a), b - trunc_bf(b));
    }
    *(u32x4*)dst_hi = ph;
    *(u32x4*)dst_lo = pl;
}

__device__ __forceinline__ void stage4(f32x4 v0,
                                       unsigned short* dst_hi,
                                       unsigned short* dst_lo) {
    u32x2 ph, pl;
#pragma unroll
    for (int p = 0; p < 2; ++p) {
        float a = v0[2 * p], b = v0[2 * p + 1];
        ph[p] = pack_hi16(a, b);
        pl[p] = pack_hi16(a - trunc_bf(a), b - trunc_bf(b));
    }
    *(u32x2*)dst_hi = ph;
    *(u32x2*)dst_lo = pl;
}

__global__ __launch_bounds__(256)
void PartialFFTConv_fir_wmma(const float* __restrict__ x,
                             const float* __restrict__ k,
                             float* __restrict__ y)
{
    __shared__ alignas(16) unsigned short xs_hi[XS_LEN];
    __shared__ alignas(16) unsigned short xs_lo[XS_LEN];
    __shared__ alignas(16) unsigned short bt_hi[16 * BT_STRIDE];
    __shared__ alignas(16) unsigned short bt_lo[16 * BT_STRIDE];
    __shared__ alignas(4)  unsigned short kh[TAPS + 4];   // [128] = zero slot
    __shared__ alignas(4)  unsigned short kl[TAPS + 4];

    const int t  = threadIdx.x;
    const int h  = blockIdx.x >> 1;       // head
    const int c  = blockIdx.x & 1;        // L-half
    const int s0 = c * SEG;               // segment start in L

    // ---- convert filter taps once (truncation hi/lo split) ----
    if (t <= TAPS) {
        float f = (t < TAPS) ? k[(long)h * LQ + t] : 0.0f;
        unsigned int u = __float_as_uint(f);
        float d = f - trunc_bf(f);
        kh[t] = (unsigned short)(u >> 16);
        kl[t] = (unsigned short)(__float_as_uint(d) >> 16);
    }
    // ---- zero pads: causal left pad (first half only) + right guard ----
    if (c == 0 && t < 64) {
        xs_hi[2 * t] = 0; xs_hi[2 * t + 1] = 0;
        xs_lo[2 * t] = 0; xs_lo[2 * t + 1] = 0;
    }
    if (t < 32) { int i = 128 + SEG + t; xs_hi[i] = 0; xs_lo[i] = 0; }
    __syncthreads();   // kh/kl ready

    // ---- build transposed Toeplitz bt[n][u'] by LDS gather (no converts) ----
    for (int e = t; e < 16 * BT_STRIDE; e += 256) {
        int n   = e / BT_STRIDE;
        int up  = e % BT_STRIDE;
        int tap = n - up + 128;
        int idx = (up < UPAD && tap >= 0 && tap < TAPS) ? tap : TAPS;  // TAPS -> 0.0
        bt_hi[e] = kh[idx];
        bt_lo[e] = kl[idx];
    }
    __syncthreads();   // bt ready

    const int lane = t & 31;              // wave32 (gfx1250)
    const int wave = t >> 5;              // 0..7 -> tile id
    const int nn   = lane & 15;
    const int kofs = (lane < 16) ? 0 : 8;
    const int mbase = (lane < 16) ? 0 : 8;

    // ---- hoist B fragments to registers: reused for 8 batches x 8 tiles ----
    V16 Bh[5], Bl[5];
#pragma unroll
    for (int j = 0; j < 5; ++j) {
        int base = nn * BT_STRIDE + 32 * j + kofs;
        Bh[j].q[0] = *(const u32x4*)&bt_hi[base];
        Bh[j].q[1] = *(const u32x4*)&bt_hi[base + 16];
        Bl[j].q[0] = *(const u32x4*)&bt_lo[base];
        Bl[j].q[1] = *(const u32x4*)&bt_lo[base + 16];
    }

    // ---- register-prefetch x segment for batch 0 ----
    const float* xr0 = x + (long)h * LQ + s0;
    f32x4 cur0 = __builtin_nontemporal_load(&((const f32x4*)xr0)[2 * t]);
    f32x4 cur1 = __builtin_nontemporal_load(&((const f32x4*)xr0)[2 * t + 1]);
    f32x4 curh = {};
    if (c == 1 && t < 32)
        curh = __builtin_nontemporal_load((const f32x4*)(xr0 - 128) + t);

    for (int b = 0; b < 8; ++b) {
        if (b) __syncthreads();           // previous batch done reading xs

        // stage current batch into LDS (packed b128/b64 stores)
        stage8(cur0, cur1, &xs_hi[128 + 8 * t], &xs_lo[128 + 8 * t]);
        if (c == 1 && t < 32)
            stage4(curh, &xs_hi[4 * t], &xs_lo[4 * t]);   // left halo
        __syncthreads();                  // xs visible

        // register-prefetch next batch behind this batch's WMMA work
        if (b < 7) {
            const float* xrn = x + ((long)(b + 1) * HD + h) * LQ + s0;
            cur0 = __builtin_nontemporal_load(&((const f32x4*)xrn)[2 * t]);
            cur1 = __builtin_nontemporal_load(&((const f32x4*)xrn)[2 * t + 1]);
            if (c == 1 && t < 32)
                curh = __builtin_nontemporal_load((const f32x4*)(xrn - 128) + t);
        }

        // ---- compute this wave's 256-output tile (pipelined A fragments) ----
        const int l0    = wave * 256;     // relative to segment
        const int abase = l0 + 16 * nn + kofs;

        V16 Ah[2], Al[2];
        Ah[0].q[0] = *(const u32x4*)&xs_hi[abase];
        Ah[0].q[1] = *(const u32x4*)&xs_hi[abase + 16];
        Al[0].q[0] = *(const u32x4*)&xs_lo[abase];
        Al[0].q[1] = *(const u32x4*)&xs_lo[abase + 16];

        v8f acc = {};
#pragma unroll
        for (int j = 0; j < 5; ++j) {
            const int cb = j & 1, nb = cb ^ 1;
            if (j < 4) {                  // chunk j+1 loads fly under chunk j's WMMAs
                int ab = abase + 32 * (j + 1);
                Ah[nb].q[0] = *(const u32x4*)&xs_hi[ab];
                Ah[nb].q[1] = *(const u32x4*)&xs_hi[ab + 16];
                Al[nb].q[0] = *(const u32x4*)&xs_lo[ab];
                Al[nb].q[1] = *(const u32x4*)&xs_lo[ab + 16];
            }
            acc = __builtin_amdgcn_wmma_f32_16x16x32_bf16(false, Ah[cb].bf, false, Bh[j].bf,
                                                          (short)0, acc, false, false);
            acc = __builtin_amdgcn_wmma_f32_16x16x32_bf16(false, Ah[cb].bf, false, Bl[j].bf,
                                                          (short)0, acc, false, false);
            acc = __builtin_amdgcn_wmma_f32_16x16x32_bf16(false, Al[cb].bf, false, Bh[j].bf,
                                                          (short)0, acc, false, false);
        }

        // D layout: VGPR r -> M=r (lanes 0-15) / M=r+8 (lanes 16-31), N=lane%16
        float* yr = y + ((long)b * HD + h) * LQ + s0;
#pragma unroll
        for (int r = 0; r < 8; ++r)
            __builtin_nontemporal_store(acc[r], &yr[l0 + 16 * (mbase + r) + nn]);
    }
}

extern "C" void kernel_launch(void* const* d_in, const int* in_sizes, int n_in,
                              void* d_out, int out_size, void* d_ws, size_t ws_size,
                              hipStream_t stream) {
    const float* x = (const float*)d_in[0];   // [8, 768, 4096] f32
    const float* k = (const float*)d_in[1];   // [768, 4096] f32 (taps 0..127 used)
    float*       y = (float*)d_out;           // [8, 768, 4096] f32

    dim3 grid(HD * 2);   // one block per (head, L-half)
    dim3 block(256);     // 8 wave32 waves; wave w owns tile w of each batch
    hipLaunchKernelGGL(PartialFFTConv_fir_wmma, grid, block, 0, stream, x, k, y);
}
